// MultiHeadAttention_87110526697836
// MI455X (gfx1250) — matrix operand order repaired and checked by reading, not compiled, not run on hardware
//
#include <hip/hip_runtime.h>
#include <hip/hip_bf16.h>

typedef __attribute__((ext_vector_type(16))) __bf16 v16bf;
typedef __attribute__((ext_vector_type(8)))  __bf16 v8bf;
typedef __attribute__((ext_vector_type(8)))  float  v8f;

constexpr int NH  = 8;      // heads
constexpr int NB  = 8;      // batch
constexpr int TT  = 1024;   // seq len
constexpr int UU  = 512;    // units
constexpr int DHD = 64;     // head dim
constexpr int HBC = 64;     // NH*NB
#define NEGV (-4294967295.0f)
#define LNEPS 1e-3f

#if __has_builtin(__builtin_amdgcn_global_load_async_to_lds_b32) && \
    __has_builtin(__builtin_amdgcn_s_wait_asynccnt)
#define HAVE_ASYNC_LDS 1
#endif

#define ASG __attribute__((address_space(1)))
#define ASL __attribute__((address_space(3)))

#ifdef HAVE_ASYNC_LDS
// Value-preserving int->pointer casts: AS1 pointers are the 64-bit global
// address; AS3 pointers are the low 32 bits of the generic LDS address
// (exactly the hardware generic->LDS truncation, ISA 10.2).
static __device__ __forceinline__ ASG int* to_global(const void* p) {
  return (ASG int*)(unsigned long long)p;
}
static __device__ __forceinline__ ASL int* to_lds(const void* p) {
  return (ASL int*)(unsigned int)(unsigned long long)p;
}
#endif

static __device__ __forceinline__ v8f wmma_bf16(v16bf a, v16bf b, v8f c) {
  return __builtin_amdgcn_wmma_f32_16x16x32_bf16(false, a, false, b, (short)0, c, false, false);
}

// ---------------- elementwise converters ----------------
__global__ void cvt_bf16_kernel(const float* __restrict__ x, __bf16* __restrict__ y, int n) {
  int i = blockIdx.x * 256 + threadIdx.x;
  if (i < n) y[i] = (__bf16)x[i];
}

// Wt[n][k] = (bf16) W[k][n]
__global__ void wtr_bf16_kernel(const float* __restrict__ w, __bf16* __restrict__ wt) {
  int i = blockIdx.x * 256 + threadIdx.x;
  if (i >= UU * UU) return;
  int k = i / UU, n = i % UU;
  wt[n * UU + k] = (__bf16)w[i];
}

// R[hb][t][j] = sum_d qh[hb][t][d] * pe_k[j][d]
__global__ void relk_kernel(const __bf16* __restrict__ qh, const float* __restrict__ pe_k,
                            float* __restrict__ R) {
  int idx = blockIdx.x * 256 + threadIdx.x;
  if (idx >= HBC * TT * 33) return;
  int j = idx % 33;
  int r = idx / 33;
  const __bf16* q = qh + (long)r * DHD;
  const float*  p = pe_k + j * DHD;
  float s = 0.f;
#pragma unroll
  for (int d = 0; d < DHD; ++d) s += (float)q[d] * p[d];
  R[idx] = s;
}

// ---------------- projection GEMM: E = relu(X @ W + b), bf16 out ----------------
// mode 0: out[hb][t][d]   (head-major, for Q and K)
// mode 1: out[hb][d][t]   (transposed, for V)
__global__ __launch_bounds__(128)
void proj_kernel(const __bf16* __restrict__ X, const __bf16* __restrict__ Wt,
                 const float* __restrict__ bias, __bf16* __restrict__ out, int mode) {
  int wid  = threadIdx.x >> 5;
  int lane = threadIdx.x & 31;
  int lo = lane & 15, hi = lane >> 4;
  int tile = blockIdx.x * 4 + wid;   // 4096 wave-tiles
  int rt = tile >> 3;                // 512 row-tiles of 16
  int ng = tile & 7;                 // 8 col-groups of 64

  int row = rt * 16 + lo;
  v8f acc[4] = {};
  const __bf16* ap = X + (long)row * UU + hi * 16;
  const __bf16* wp = Wt + (long)(ng * 64 + lo) * UU + hi * 16;

  auto loadA = [&](int ks, v16bf& a) {
    v8bf alo = *(const v8bf*)(ap + ks * 32);
    v8bf ahi = *(const v8bf*)(ap + ks * 32 + 8);
#pragma unroll
    for (int e = 0; e < 8; ++e) { a[e] = alo[e]; a[e + 8] = ahi[e]; }
  };
  auto loadB = [&](int ks, v16bf* b) {
#pragma unroll
    for (int dt = 0; dt < 4; ++dt)
      b[dt] = *(const v16bf*)(wp + (long)dt * 16 * UU + ks * 32);
  };

  v16bf a0, a1, b0[4], b1[4];
  loadA(0, a0); loadB(0, b0);
#pragma unroll
  for (int ks = 0; ks < 16; ks += 2) {
    loadA(ks + 1, a1); loadB(ks + 1, b1);   // next step in flight
#pragma unroll
    for (int dt = 0; dt < 4; ++dt) acc[dt] = wmma_bf16(a0, b0[dt], acc[dt]);
    if (ks + 2 < 16) { loadA(ks + 2, a0); loadB(ks + 2, b0); }
#pragma unroll
    for (int dt = 0; dt < 4; ++dt) acc[dt] = wmma_bf16(a1, b1[dt], acc[dt]);
  }

#pragma unroll
  for (int dt = 0; dt < 4; ++dt) {
    int n = ng * 64 + dt * 16 + lo;
    float bv = bias[n];
#pragma unroll
    for (int i = 0; i < 8; ++i) {
      int r = rt * 16 + i + hi * 8;
      float v = acc[dt][i] + bv;
      v = v > 0.f ? v : 0.f;
      int bb = r >> 10, t = r & 1023;
      int h = n >> 6, d = n & 63;
      int hb = h * NB + bb;
      long idx = (mode == 0) ? ((long)(hb * TT + t)) * DHD + d
                             : ((long)(hb * DHD + d)) * TT + t;
      out[idx] = (__bf16)v;
    }
  }
}

// ---------------- flash attention with relative positions ----------------
__global__ __launch_bounds__(128)
void attn_kernel(const __bf16* __restrict__ qh, const __bf16* __restrict__ kh,
                 const __bf16* __restrict__ vT, const float* __restrict__ R,
                 const float* __restrict__ pe_v, __bf16* __restrict__ obuf) {
  __shared__ float pev[17 * 64];
  __shared__ float Rbuf[4][16 * 33];
  __shared__ __align__(16) __bf16 pbuf[4][16 * 32];
  __shared__ float band[4][16 * 16];

  int wid = threadIdx.x >> 5, lane = threadIdx.x & 31;
  int lo = lane & 15, hi = lane >> 4;
  int tile = blockIdx.x * 4 + wid;   // 4096 = 64 hb * 64 t-tiles
  int hb = tile >> 6;
  int t0 = (tile & 63) * 16;

#ifdef HAVE_ASYNC_LDS
  for (int i = threadIdx.x; i < 17 * 64; i += 128)
    __builtin_amdgcn_global_load_async_to_lds_b32(to_global(pe_v + i), to_lds(&pev[i]), 0, 0);
  for (int i = lane; i < 16 * 33; i += 32)
    __builtin_amdgcn_global_load_async_to_lds_b32(
        to_global(R + ((long)hb * TT + t0) * 33 + i), to_lds(&Rbuf[wid][i]), 0, 0);
#else
  for (int i = threadIdx.x; i < 17 * 64; i += 128) pev[i] = pe_v[i];
  for (int i = lane; i < 16 * 33; i += 32)
    Rbuf[wid][i] = R[((long)hb * TT + t0) * 33 + i];
#endif
  for (int i = lane; i < 256; i += 32) band[wid][i] = -1e30f;
#ifdef HAVE_ASYNC_LDS
  __builtin_amdgcn_s_wait_asynccnt(0);
#endif
  __syncthreads();

  // Q fragments (16 rows x 64 d) as two 16x32 A fragments
  int trow = t0 + lo;
  const __bf16* qp = qh + ((long)hb * TT + trow) * DHD + hi * 16;
  v16bf aq0, aq1;
  {
    v8bf l0 = *(const v8bf*)(qp);      v8bf h0 = *(const v8bf*)(qp + 8);
    v8bf l1 = *(const v8bf*)(qp + 32); v8bf h1 = *(const v8bf*)(qp + 40);
#pragma unroll
    for (int e = 0; e < 8; ++e) {
      aq0[e] = l0[e]; aq0[e + 8] = h0[e];
      aq1[e] = l1[e]; aq1[e + 8] = h1[e];
    }
  }

  float r0reg[8];
#pragma unroll
  for (int i = 0; i < 8; ++i) r0reg[i] = Rbuf[wid][(i + hi * 8) * 33];

  v8f o[4] = {};
  float mrun[8], lrun[8];
#pragma unroll
  for (int i = 0; i < 8; ++i) { mrun[i] = -1e30f; lrun[i] = 0.f; }

  auto loadK = [&](int s0, v16bf* bk) {
    const __bf16* kp = kh + ((long)hb * TT + s0 + lo) * DHD;
    bk[0] = *(const v16bf*)(kp + hi * 16);
    bk[1] = *(const v16bf*)(kp + 32 + hi * 16);
    bk[2] = *(const v16bf*)(kp + 16 * DHD + hi * 16);
    bk[3] = *(const v16bf*)(kp + 16 * DHD + 32 + hi * 16);
  };

  v16bf bk[4], nk[4], bv[4];
  loadK(0, bk);
#pragma unroll
  for (int dt = 0; dt < 4; ++dt) nk[dt] = bk[dt];

  int nsb = (t0 + 15) / 32 + 1;
  for (int sb = 0; sb < nsb; ++sb) {
    int s0 = sb * 32;
    // V fragments for this block: issued early, consumed after softmax
    const __bf16* vp = vT + (long)hb * DHD * TT + s0 + hi * 16;
#pragma unroll
    for (int dt = 0; dt < 4; ++dt)
      bv[dt] = *(const v16bf*)(vp + (long)(dt * 16 + lo) * TT);
    if (sb + 1 < nsb) loadK(s0 + 32, nk);     // K double buffer
    if (sb + 2 < nsb) {                        // distance-2 prefetch
      __builtin_prefetch(kh + ((long)hb * TT + s0 + 64 + lo) * DHD, 0, 1);
      __builtin_prefetch(vT + (long)hb * DHD * TT + (long)lo * TT + s0 + 64, 0, 1);
    }

    v8f S0 = {}, S1 = {};
    S0 = wmma_bf16(aq0, bk[0], S0);
    S0 = wmma_bf16(aq1, bk[1], S0);
    S1 = wmma_bf16(aq0, bk[2], S1);
    S1 = wmma_bf16(aq1, bk[3], S1);

    // relative-position logit bias
    if (s0 + 47 <= t0) {           // whole block clips to j=0
#pragma unroll
      for (int i = 0; i < 8; ++i) { S0[i] += r0reg[i]; S1[i] += r0reg[i]; }
    } else {
#pragma unroll
      for (int i = 0; i < 8; ++i) {
        int ri = i + hi * 8, t = t0 + ri;
        int j0 = s0 + lo - t;      j0 = j0 < -16 ? -16 : (j0 > 16 ? 16 : j0);
        int j1 = s0 + 16 + lo - t; j1 = j1 < -16 ? -16 : (j1 > 16 ? 16 : j1);
        S0[i] += Rbuf[wid][ri * 33 + j0 + 16];
        S1[i] += Rbuf[wid][ri * 33 + j1 + 16];
      }
    }
#pragma unroll
    for (int i = 0; i < 8; ++i) { S0[i] *= 0.125f; S1[i] *= 0.125f; }
    if (s0 + 31 > t0) {            // causal mask on diagonal blocks
#pragma unroll
      for (int i = 0; i < 8; ++i) {
        int t = t0 + i + hi * 8;
        if (s0 + lo > t)      S0[i] = NEGV;
        if (s0 + 16 + lo > t) S1[i] = NEGV;
      }
    }
    if (s0 + 47 > t0) {            // capture near-diagonal logits (k = t-s in [0,16))
#pragma unroll
      for (int i = 0; i < 8; ++i) {
        int ri = i + hi * 8, t = t0 + ri;
        int k0 = t - (s0 + lo);
        if (k0 >= 0 && k0 < 16) band[wid][ri * 16 + k0] = S0[i];
        int k1 = t - (s0 + 16 + lo);
        if (k1 >= 0 && k1 < 16) band[wid][ri * 16 + k1] = S1[i];
      }
    }
    // online softmax
    float fac[8];
#pragma unroll
    for (int i = 0; i < 8; ++i) {
      float v = fmaxf(S0[i], S1[i]);
      v = fmaxf(v, __shfl_xor(v, 1, 16));
      v = fmaxf(v, __shfl_xor(v, 2, 16));
      v = fmaxf(v, __shfl_xor(v, 4, 16));
      v = fmaxf(v, __shfl_xor(v, 8, 16));
      float mnew = fmaxf(mrun[i], v);
      fac[i] = __expf(mrun[i] - mnew);
      mrun[i] = mnew;
      S0[i] = __expf(S0[i] - mnew);
      S1[i] = __expf(S1[i] - mnew);
      float rs = S0[i] + S1[i];
      rs += __shfl_xor(rs, 1, 16);
      rs += __shfl_xor(rs, 2, 16);
      rs += __shfl_xor(rs, 4, 16);
      rs += __shfl_xor(rs, 8, 16);
      lrun[i] = lrun[i] * fac[i] + rs;
    }
#pragma unroll
    for (int dt = 0; dt < 4; ++dt)
#pragma unroll
      for (int i = 0; i < 8; ++i) o[dt][i] *= fac[i];
    // transpose P (C-layout -> A-fragment) through LDS
#pragma unroll
    for (int i = 0; i < 8; ++i) {
      int ri = i + hi * 8;
      pbuf[wid][ri * 32 + lo]      = (__bf16)S0[i];
      pbuf[wid][ri * 32 + 16 + lo] = (__bf16)S1[i];
    }
    v16bf ap;
    {
      const __bf16* pp = &pbuf[wid][lo * 32 + hi * 16];
      v8bf plo = *(const v8bf*)(pp);
      v8bf phi = *(const v8bf*)(pp + 8);
#pragma unroll
      for (int e = 0; e < 8; ++e) { ap[e] = plo[e]; ap[e + 8] = phi[e]; }
    }
    // O += P @ V
#pragma unroll
    for (int dt = 0; dt < 4; ++dt) o[dt] = wmma_bf16(ap, bv[dt], o[dt]);
#pragma unroll
    for (int dt = 0; dt < 4; ++dt) bk[dt] = nk[dt];
  }

  // epilogue: normalize, rel_v contribution, store
  float inv[8];
#pragma unroll
  for (int i = 0; i < 8; ++i) inv[i] = 1.0f / lrun[i];
#pragma unroll
  for (int dt = 0; dt < 4; ++dt) {
    float pv0 = pev[dt * 16 + lo];
#pragma unroll
    for (int i = 0; i < 8; ++i) o[dt][i] = o[dt][i] * inv[i] + pv0;
  }
  for (int k = 0; k < 16; ++k) {
    float pv[4];
#pragma unroll
    for (int dt = 0; dt < 4; ++dt)
      pv[dt] = pev[(16 - k) * 64 + dt * 16 + lo] - pev[dt * 16 + lo];
#pragma unroll
    for (int i = 0; i < 8; ++i) {
      int ri = i + hi * 8;
      float ew = __expf(band[wid][ri * 16 + k] - mrun[i]) * inv[i];
#pragma unroll
      for (int dt = 0; dt < 4; ++dt) o[dt][i] += ew * pv[dt];
    }
  }
  int h = hb >> 3, bb = hb & 7;
#pragma unroll
  for (int i = 0; i < 8; ++i) {
    int t = t0 + i + hi * 8;
    long base = (((long)bb * TT + t) * NH + h) * DHD;
#pragma unroll
    for (int dt = 0; dt < 4; ++dt) obuf[base + dt * 16 + lo] = (__bf16)o[dt][i];
  }
}

// ---------------- output projection + residual + LayerNorm ----------------
__global__ __launch_bounds__(128)
void outln_kernel(const __bf16* __restrict__ X, const __bf16* __restrict__ Wot,
                  const float* __restrict__ bo, const float* __restrict__ qres,
                  const float* __restrict__ gamma, const float* __restrict__ beta,
                  float* __restrict__ out) {
  __shared__ float ssum[16][4];
  __shared__ float ssq[16][4];
  int wid = threadIdx.x >> 5, lane = threadIdx.x & 31;
  int lo = lane & 15, hi = lane >> 4;
  int rt = blockIdx.x;               // 512 blocks, 16 rows each
  int row = rt * 16 + lo;

  v8f acc[8] = {};
  const __bf16* ap = X + (long)row * UU + hi * 16;
  const __bf16* wp = Wot + (long)(wid * 128 + lo) * UU + hi * 16;

  auto loadA = [&](int ks, v16bf& a) {
    v8bf alo = *(const v8bf*)(ap + ks * 32);
    v8bf ahi = *(const v8bf*)(ap + ks * 32 + 8);
#pragma unroll
    for (int e = 0; e < 8; ++e) { a[e] = alo[e]; a[e + 8] = ahi[e]; }
  };
  auto loadB = [&](int ks, v16bf* b) {
#pragma unroll
    for (int nt = 0; nt < 8; ++nt)
      b[nt] = *(const v16bf*)(wp + (long)nt * 16 * UU + ks * 32);
  };

  v16bf a0, a1, b0[8], b1[8];
  loadA(0, a0); loadB(0, b0);
#pragma unroll 2
  for (int ks = 0; ks < 16; ks += 2) {
    loadA(ks + 1, a1); loadB(ks + 1, b1);
#pragma unroll
    for (int nt = 0; nt < 8; ++nt) acc[nt] = wmma_bf16(a0, b0[nt], acc[nt]);
    if (ks + 2 < 16) { loadA(ks + 2, a0); loadB(ks + 2, b0); }
#pragma unroll
    for (int nt = 0; nt < 8; ++nt) acc[nt] = wmma_bf16(a1, b1[nt], acc[nt]);
  }

  float vals[8][8];
  float psum[8], psq[8];
#pragma unroll
  for (int i = 0; i < 8; ++i) { psum[i] = 0.f; psq[i] = 0.f; }
#pragma unroll
  for (int nt = 0; nt < 8; ++nt) {
    int n = wid * 128 + nt * 16 + lo;
    float bv = bo[n];
#pragma unroll
    for (int i = 0; i < 8; ++i) {
      int r = rt * 16 + i + hi * 8;
      float v = acc[nt][i] + bv;
      v = v > 0.f ? v : 0.f;
      v += qres[(long)r * UU + n];
      vals[nt][i] = v;
      psum[i] += v; psq[i] += v * v;
    }
  }
#pragma unroll
  for (int i = 0; i < 8; ++i) {
    float s = psum[i], q2 = psq[i];
    s  += __shfl_xor(s, 1, 16);  s  += __shfl_xor(s, 2, 16);
    s  += __shfl_xor(s, 4, 16);  s  += __shfl_xor(s, 8, 16);
    q2 += __shfl_xor(q2, 1, 16); q2 += __shfl_xor(q2, 2, 16);
    q2 += __shfl_xor(q2, 4, 16); q2 += __shfl_xor(q2, 8, 16);
    psum[i] = s; psq[i] = q2;
  }
  if (lo == 0) {
#pragma unroll
    for (int i = 0; i < 8; ++i) {
      ssum[i + hi * 8][wid] = psum[i];
      ssq[i + hi * 8][wid]  = psq[i];
    }
  }
  __syncthreads();
#pragma unroll
  for (int i = 0; i < 8; ++i) {
    int ri = i + hi * 8;
    float s  = ssum[ri][0] + ssum[ri][1] + ssum[ri][2] + ssum[ri][3];
    float q2 = ssq[ri][0]  + ssq[ri][1]  + ssq[ri][2]  + ssq[ri][3];
    float mu = s * (1.0f / UU);
    float var = q2 * (1.0f / UU) - mu * mu;
    float rstd = rsqrtf(var + LNEPS);
    int r = rt * 16 + ri;
#pragma unroll
    for (int nt = 0; nt < 8; ++nt) {
      int n = wid * 128 + nt * 16 + lo;
      out[(long)r * UU + n] = (vals[nt][i] - mu) * rstd * gamma[n] + beta[n];
    }
  }
}

extern "C" void kernel_launch(void* const* d_in, const int* in_sizes, int n_in,
                              void* d_out, int out_size, void* d_ws, size_t ws_size,
                              hipStream_t stream) {
  const float* q     = (const float*)d_in[0];
  const float* k     = (const float*)d_in[1];
  const float* v     = (const float*)d_in[2];
  const float* Wq    = (const float*)d_in[3];
  const float* bq    = (const float*)d_in[4];
  const float* Wk    = (const float*)d_in[5];
  const float* bk    = (const float*)d_in[6];
  const float* Wv    = (const float*)d_in[7];
  const float* bv    = (const float*)d_in[8];
  const float* Wo    = (const float*)d_in[9];
  const float* bo    = (const float*)d_in[10];
  const float* gamma = (const float*)d_in[11];
  const float* beta  = (const float*)d_in[12];
  const float* pe_k  = (const float*)d_in[13];
  const float* pe_v  = (const float*)d_in[14];
  float* out = (float*)d_out;

  char* ws = (char*)d_ws;
  size_t off = 0;
  auto alloc = [&](size_t bytes) {
    void* p = ws + off;
    off += (bytes + 255) & ~(size_t)255;
    return p;
  };
  const size_t NTOK = (size_t)NB * TT;              // 8192
  __bf16* qb  = (__bf16*)alloc(NTOK * UU * 2);      // bf16 inputs
  __bf16* kb  = (__bf16*)alloc(NTOK * UU * 2);
  __bf16* vb  = (__bf16*)alloc(NTOK * UU * 2);
  __bf16* Wqt = (__bf16*)alloc((size_t)UU * UU * 2);
  __bf16* Wkt = (__bf16*)alloc((size_t)UU * UU * 2);
  __bf16* Wvt = (__bf16*)alloc((size_t)UU * UU * 2);
  __bf16* Wot = (__bf16*)alloc((size_t)UU * UU * 2);
  __bf16* qhb = (__bf16*)alloc((size_t)HBC * TT * DHD * 2);
  __bf16* khb = (__bf16*)alloc((size_t)HBC * TT * DHD * 2);
  __bf16* vTb = (__bf16*)alloc((size_t)HBC * DHD * TT * 2);
  // reuse regions freed after the projections:
  float*  R    = (float*)qb;      // needs 8.65 MB, qb+kb region free after projections
  __bf16* obuf = vb;              // needs 8.39 MB, vb free after V projection

  int ncvt = (int)(NTOK * UU);
  cvt_bf16_kernel<<<(ncvt + 255) / 256, 256, 0, stream>>>(q, qb, ncvt);
  cvt_bf16_kernel<<<(ncvt + 255) / 256, 256, 0, stream>>>(k, kb, ncvt);
  cvt_bf16_kernel<<<(ncvt + 255) / 256, 256, 0, stream>>>(v, vb, ncvt);
  int nwt = UU * UU;
  wtr_bf16_kernel<<<(nwt + 255) / 256, 256, 0, stream>>>(Wq, Wqt);
  wtr_bf16_kernel<<<(nwt + 255) / 256, 256, 0, stream>>>(Wk, Wkt);
  wtr_bf16_kernel<<<(nwt + 255) / 256, 256, 0, stream>>>(Wv, Wvt);
  wtr_bf16_kernel<<<(nwt + 255) / 256, 256, 0, stream>>>(Wo, Wot);

  proj_kernel<<<1024, 128, 0, stream>>>(qb, Wqt, bq, qhb, 0);
  proj_kernel<<<1024, 128, 0, stream>>>(kb, Wkt, bk, khb, 0);
  proj_kernel<<<1024, 128, 0, stream>>>(vb, Wvt, bv, vTb, 1);

  int nrel = HBC * TT * 33;
  relk_kernel<<<(nrel + 255) / 256, 256, 0, stream>>>(qhb, pe_k, R);

  attn_kernel<<<1024, 128, 0, stream>>>(qhb, khb, vTb, R, pe_v, obuf);

  outln_kernel<<<512, 128, 0, stream>>>(obuf, Wot, bo, q, gamma, beta, out);
}
